// LSTM_AE_GMM_10256381903632
// MI455X (gfx1250) — matrix operand order repaired and checked by reading, not compiled, not run on hardware
//
#include <hip/hip_runtime.h>
#include <hip/hip_bf16.h>

// Model dims (fixed by the reference)
#define BB 128
#define TT 256
#define HH 64
#define VV 1024
#define BT (BB*TT)    // 32768
#define G3 192        // 3*H

typedef __attribute__((ext_vector_type(16))) __bf16 v16bf;
typedef __attribute__((ext_vector_type(8)))  __bf16 v8bf;
typedef __attribute__((ext_vector_type(8)))  float  v8f;

union Frag16 { v16bf v; v8bf h[2]; unsigned short e[16]; };

__device__ __forceinline__ unsigned short f2bf(float f) {
  unsigned u = __float_as_uint(f);
  u += 0x7fffu + ((u >> 16) & 1u);          // round-to-nearest-even
  return (unsigned short)(u >> 16);
}

__device__ __forceinline__ float sigmoidf_(float x) { return 1.f / (1.f + __expf(-x)); }

// A operand: 16x32 bf16 tile, row-major source [.., lda], rows = mbase..mbase+15
// lane<16: K in [0,8)+[16,24); lane>=16: K in [8,16)+[24,32)  (ISA 7.12.2)
__device__ __forceinline__ v16bf load_a_frag(const unsigned short* A, int lda,
                                             int mbase, int kbase, int lane) {
  int hf = lane >> 4;
  const unsigned short* p = A + (size_t)(mbase + (lane & 15)) * lda + kbase + hf * 8;
  Frag16 f;
  f.h[0] = *(const v8bf*)p;
  f.h[1] = *(const v8bf*)(p + 16);
  return f.v;
}

// B operand: 32x16 bf16 tile from W stored [N,K] row-major.
// lane = output column; lane<16: K in [0,8)+[16,24); lane>=16: K in [8,16)+[24,32)
__device__ __forceinline__ v16bf load_b_frag(const unsigned short* W, int ldw,
                                             int nbase, int kbase, int lane) {
  int hf = lane >> 4;
  const unsigned short* p = W + (size_t)(nbase + (lane & 15)) * ldw + kbase + hf * 8;
  Frag16 f;
  f.h[0] = *(const v8bf*)p;
  f.h[1] = *(const v8bf*)(p + 16);
  return f.v;
}

__device__ __forceinline__ v8f wmma_bf16(v16bf a, v16bf b, v8f c) {
  return __builtin_amdgcn_wmma_f32_16x16x32_bf16(false, a, false, b, (short)0, c, false, false);
}

// ---------------------------------------------------------------- cvt f32->bf16
__global__ void cvt_kernel(const float* __restrict__ s, unsigned short* __restrict__ d, int n) {
  int i = blockIdx.x * blockDim.x + threadIdx.x;
  if (i < n) d[i] = f2bf(s[i]);
}

// ---------------------------------------------------------------- embedding gather -> bf16
__global__ void embed_kernel(const int* __restrict__ x, const float* __restrict__ emb,
                             unsigned short* __restrict__ ex, int n) {
  int idx = blockIdx.x * blockDim.x + threadIdx.x;
  if (idx < n) {
    int bt = idx >> 6, e = idx & 63;
    ex[idx] = f2bf(emb[(size_t)x[bt] * 64 + e]);
  }
}

// ---------------------------------------------------------------- generic Y = A @ W^T + bias
// A: bf16 [M, lda] (K used cols), W: bf16 [N, K], bias fp32[N]
// mode 0: outF fp32 [M, ldo];  mode 1: SELU then bf16 to outB [M, ldo]
__global__ void gemm_bias_kernel(const unsigned short* __restrict__ A, int lda,
                                 const unsigned short* __restrict__ W,
                                 const float* __restrict__ bias,
                                 float* __restrict__ outF, unsigned short* __restrict__ outB,
                                 int ldo, int M, int N, int K, int mode) {
  int lane = threadIdx.x & 31;
  int wave = threadIdx.x >> 5;
  int wpb  = blockDim.x >> 5;
  int gw   = blockIdx.x * wpb + wave;
  int nw   = gridDim.x * wpb;
  int mstr = M >> 4, nstr = N >> 6;          // 16-row x 64-col strips per wave
  int total = mstr * nstr;
  for (int sid = gw; sid < total; sid += nw) {
    int mi = sid / nstr, ni = sid % nstr;
    v8f acc[4];
#pragma unroll
    for (int n4 = 0; n4 < 4; ++n4)
#pragma unroll
      for (int i = 0; i < 8; ++i) acc[n4][i] = 0.f;

    for (int kc = 0; kc < K; kc += 32) {
      v16bf a = load_a_frag(A, lda, mi * 16, kc, lane);
#pragma unroll
      for (int n4 = 0; n4 < 4; ++n4) {
        v16bf b = load_b_frag(W, K, ni * 64 + n4 * 16, kc, lane);
        acc[n4] = wmma_bf16(a, b, acc[n4]);
      }
    }
    int hf = lane >> 4, cr = lane & 15;
#pragma unroll
    for (int n4 = 0; n4 < 4; ++n4) {
      int col = ni * 64 + n4 * 16 + cr;
      float bv = bias ? bias[col] : 0.f;
#pragma unroll
      for (int i = 0; i < 8; ++i) {
        int row = mi * 16 + i + hf * 8;
        float v = acc[n4][i] + bv;
        if (mode == 1) {
          v = 1.0507009873554805f * (v > 0.f ? v : 1.6732632423543772f * (__expf(v) - 1.f));
          outB[(size_t)row * ldo + col] = f2bf(v);
        } else {
          outF[(size_t)row * ldo + col] = v;
        }
      }
    }
  }
}

// ---------------------------------------------------------------- GRU recurrence (one biGRU layer)
// grid = (8 b-blocks, 2 dirs), blockDim = 128 (4 waves). Each wave owns 16 hidden cols.
// xi: fp32 gate preacts incl. bih; strides parameterized (t-stride 0 => broadcast decoder input)
// whh: bf16 [2,192,64]; bhh fp32 [2,192]
// y (optional): bf16 output at [ (b*T+ti)*ldy + ycol + dir*64 + j ]
// hout (optional): final hidden bf16 at [ b*hld + hcol + dir*64 + j ]
__global__ void gru_rec_kernel(const float* __restrict__ xi, long xi_dir_stride,
                               int xi_b_stride, int xi_t_stride,
                               const unsigned short* __restrict__ whh,
                               const float* __restrict__ bhh,
                               unsigned short* y, int ldy, int ycol,
                               unsigned short* hout, int hld, int hcol) {
  __shared__ float hs[16][65];                       // padded: bank-conflict free
  int bb   = blockIdx.x;
  int dir  = blockIdx.y;
  int lane = threadIdx.x & 31;
  int w    = threadIdx.x >> 5;                       // 0..3 -> hidden col block
  int hf   = lane >> 4, cr = lane & 15;
  int j    = w * 16 + cr;                            // hidden column 0..63

  const unsigned short* whd = whh + (size_t)dir * G3 * HH;
  const float* bhp = bhh + dir * G3;
  float bh_r = bhp[j], bh_z = bhp[64 + j], bh_n = bhp[128 + j];
  const float* xip = xi + (size_t)dir * xi_dir_stride;

  // Whh B-fragments resident in VGPRs for the whole sequence
  v16bf bfr[3][2];
#pragma unroll
  for (int g = 0; g < 3; ++g)
#pragma unroll
    for (int kc = 0; kc < 2; ++kc)
      bfr[g][kc] = load_b_frag(whd, HH, g * 64 + w * 16, kc * 32, lane);

  for (int i = threadIdx.x; i < 16 * 65; i += blockDim.x) ((float*)hs)[i] = 0.f;
  __syncthreads();

  float h2v[8];
  int rr = cr;                                        // A-operand row = lane&15
  for (int s = 0; s < TT; ++s) {
    int ti = dir ? (TT - 1 - s) : s;

    // A fragments from fp32 h in LDS (convert on the fly)
    Frag16 fa[2];
#pragma unroll
    for (int q = 0; q < 8; ++q) {
      fa[0].e[q]     = f2bf(hs[rr][hf * 8 + q]);
      fa[0].e[8 + q] = f2bf(hs[rr][16 + hf * 8 + q]);
      fa[1].e[q]     = f2bf(hs[rr][32 + hf * 8 + q]);
      fa[1].e[8 + q] = f2bf(hs[rr][48 + hf * 8 + q]);
    }
    v8f ar, az, an;
#pragma unroll
    for (int i = 0; i < 8; ++i) { ar[i] = 0.f; az[i] = 0.f; an[i] = 0.f; }
    ar = wmma_bf16(fa[0].v, bfr[0][0], ar); ar = wmma_bf16(fa[1].v, bfr[0][1], ar);
    az = wmma_bf16(fa[0].v, bfr[1][0], az); az = wmma_bf16(fa[1].v, bfr[1][1], az);
    an = wmma_bf16(fa[0].v, bfr[2][0], an); an = wmma_bf16(fa[1].v, bfr[2][1], an);

#pragma unroll
    for (int i = 0; i < 8; ++i) {
      int rl = i + hf * 8;
      int bg = bb * 16 + rl;
      const float* xp = xip + (size_t)bg * xi_b_stride + (size_t)ti * xi_t_stride;
      float xr = xp[j], xz = xp[64 + j], xn = xp[128 + j];
      float r = sigmoidf_(xr + ar[i] + bh_r);
      float z = sigmoidf_(xz + az[i] + bh_z);
      float n = tanhf(xn + r * (an[i] + bh_n));
      float h2 = (1.f - z) * n + z * hs[rl][j];
      h2v[i] = h2;
      if (y) y[((size_t)bg * TT + ti) * ldy + ycol + dir * 64 + j] = f2bf(h2);
    }
    __syncthreads();
#pragma unroll
    for (int i = 0; i < 8; ++i) hs[i + hf * 8][j] = h2v[i];
    __syncthreads();
  }

  if (hout) {
#pragma unroll
    for (int i = 0; i < 8; ++i) {
      int bg = bb * 16 + i + hf * 8;
      hout[(size_t)bg * hld + hcol + dir * 64 + j] = f2bf(h2v[i]);
    }
  }
}

// ---------------------------------------------------------------- fused fc2 + log_softmax + CE
// hdd bf16 [BT,64], W bf16 [V,64], bias fp32[V]. One block per 16 rows; 8 waves split V.
__global__ void fc2_ce_kernel(const unsigned short* __restrict__ hdd,
                              const unsigned short* __restrict__ W,
                              const float* __restrict__ bias,
                              const int* __restrict__ x, float* __restrict__ ce) {
  __shared__ float ssum[16];
  __shared__ float stgt[16];
  int mb   = blockIdx.x * 16;
  int lane = threadIdx.x & 31;
  int wave = threadIdx.x >> 5;
  int hf   = lane >> 4, cr = lane & 15;
  if (threadIdx.x < 16) { ssum[threadIdx.x] = 0.f; stgt[threadIdx.x] = 0.f; }
  __syncthreads();

  v16bf a0 = load_a_frag(hdd, HH, mb, 0, lane);
  v16bf a1 = load_a_frag(hdd, HH, mb, 32, lane);
  int tg[8];
#pragma unroll
  for (int i = 0; i < 8; ++i) tg[i] = x[mb + i + hf * 8];

  float psum[8];
#pragma unroll
  for (int i = 0; i < 8; ++i) psum[i] = 0.f;

  for (int nt = wave; nt < VV / 16; nt += 8) {
    v16bf b0 = load_b_frag(W, HH, nt * 16, 0, lane);
    v16bf b1 = load_b_frag(W, HH, nt * 16, 32, lane);
    v8f acc;
#pragma unroll
    for (int i = 0; i < 8; ++i) acc[i] = 0.f;
    acc = wmma_bf16(a0, b0, acc);
    acc = wmma_bf16(a1, b1, acc);
    int col = nt * 16 + cr;
    float bv = bias[col];
#pragma unroll
    for (int i = 0; i < 8; ++i) {
      float l = acc[i] + bv;
      psum[i] += __expf(l);
      if (col == tg[i]) stgt[i + hf * 8] = l;     // exactly one writer per row
    }
  }
#pragma unroll
  for (int i = 0; i < 8; ++i) atomicAdd(&ssum[i + hf * 8], psum[i]);
  __syncthreads();
  if (threadIdx.x < 16) ce[mb + threadIdx.x] = __logf(ssum[threadIdx.x]) - stgt[threadIdx.x];
}

// ---------------------------------------------------------------- masked mean -> scalar loss
__global__ void loss_kernel(const float* __restrict__ ce, const int* __restrict__ x,
                            float* __restrict__ out) {
  __shared__ float red[128];
  int b = threadIdx.x;                               // blockDim = 128
  float s = 0.f, c = 0.f;
  for (int t = 0; t < TT; ++t) {
    float m = (x[b * TT + t] != 0) ? 1.f : 0.f;
    s += ce[b * TT + t] * m;
    c += m;
  }
  red[b] = s / c;
  __syncthreads();
  for (int off = 64; off > 0; off >>= 1) {
    if (b < off) red[b] += red[b + off];
    __syncthreads();
  }
  if (b == 0) out[0] = red[0] / (float)BB;
}

// ================================================================ host side
extern "C" void kernel_launch(void* const* d_in, const int* in_sizes, int n_in,
                              void* d_out, int out_size, void* d_ws, size_t ws_size,
                              hipStream_t stream) {
  const int*   x     = (const int*)  d_in[0];
  const float* emb   = (const float*)d_in[1];
  const float* e0wih = (const float*)d_in[2];
  const float* e0whh = (const float*)d_in[3];
  const float* e0bih = (const float*)d_in[4];
  const float* e0bhh = (const float*)d_in[5];
  const float* e1wih = (const float*)d_in[6];
  const float* e1whh = (const float*)d_in[7];
  const float* e1bih = (const float*)d_in[8];
  const float* e1bhh = (const float*)d_in[9];
  const float* d0wih = (const float*)d_in[10];
  const float* d0whh = (const float*)d_in[11];
  const float* d0bih = (const float*)d_in[12];
  const float* d0bhh = (const float*)d_in[13];
  const float* d1wih = (const float*)d_in[14];
  const float* d1whh = (const float*)d_in[15];
  const float* d1bih = (const float*)d_in[16];
  const float* d1bhh = (const float*)d_in[17];
  const float* fc1w  = (const float*)d_in[18];
  const float* fc1b  = (const float*)d_in[19];
  const float* fc2w  = (const float*)d_in[20];
  const float* fc2b  = (const float*)d_in[21];
  float* out = (float*)d_out;

  char* ws = (char*)d_ws;
  size_t off = 0;
  auto alloc = [&](size_t bytes) -> char* {
    char* p = ws + off;
    off = (off + bytes + 255) & ~(size_t)255;
    return p;
  };
  unsigned short* wb_e0wih = (unsigned short*)alloc((size_t)2 * G3 * 64  * 2);
  unsigned short* wb_e0whh = (unsigned short*)alloc((size_t)2 * G3 * 64  * 2);
  unsigned short* wb_e1wih = (unsigned short*)alloc((size_t)2 * G3 * 128 * 2);
  unsigned short* wb_e1whh = (unsigned short*)alloc((size_t)2 * G3 * 64  * 2);
  unsigned short* wb_d0wih = (unsigned short*)alloc((size_t)2 * G3 * 256 * 2);
  unsigned short* wb_d0whh = (unsigned short*)alloc((size_t)2 * G3 * 64  * 2);
  unsigned short* wb_d1wih = (unsigned short*)alloc((size_t)2 * G3 * 128 * 2);
  unsigned short* wb_d1whh = (unsigned short*)alloc((size_t)2 * G3 * 64  * 2);
  unsigned short* wb_fc1   = (unsigned short*)alloc((size_t)64 * 256 * 2);
  unsigned short* wb_fc2   = (unsigned short*)alloc((size_t)VV * 64 * 2);
  unsigned short* ex    = (unsigned short*)alloc((size_t)BT * 64  * 2);
  unsigned short* y0    = (unsigned short*)alloc((size_t)BT * 128 * 2);
  unsigned short* res   = (unsigned short*)alloc((size_t)BT * 256 * 2);
  unsigned short* feats = (unsigned short*)alloc((size_t)BB * 256 * 2);
  unsigned short* hdd   = (unsigned short*)alloc((size_t)BT * 64  * 2);
  float* xi     = (float*)alloc((size_t)2 * BT * G3 * 4);
  float* xi_d0  = (float*)alloc((size_t)2 * BB * G3 * 4);
  float* ce     = (float*)alloc((size_t)BT * 4);

  auto cvt = [&](const float* s, unsigned short* d, int n) {
    cvt_kernel<<<(n + 255) / 256, 256, 0, stream>>>(s, d, n);
  };
  cvt(e0wih, wb_e0wih, 2 * G3 * 64);
  cvt(e0whh, wb_e0whh, 2 * G3 * 64);
  cvt(e1wih, wb_e1wih, 2 * G3 * 128);
  cvt(e1whh, wb_e1whh, 2 * G3 * 64);
  cvt(d0wih, wb_d0wih, 2 * G3 * 256);
  cvt(d0whh, wb_d0whh, 2 * G3 * 64);
  cvt(d1wih, wb_d1wih, 2 * G3 * 128);
  cvt(d1whh, wb_d1whh, 2 * G3 * 64);
  cvt(fc1w,  wb_fc1,   64 * 256);
  cvt(fc2w,  wb_fc2,   VV * 64);

  embed_kernel<<<(BT * 64 + 255) / 256, 256, 0, stream>>>(x, emb, ex, BT * 64);

  auto gemm = [&](const unsigned short* A, int lda, const unsigned short* W,
                  const float* bias, float* oF, unsigned short* oB, int ldo,
                  int M, int N, int K, int mode) {
    int strips = (M >> 4) * (N >> 6);
    int blocks = (strips + 7) / 8;
    if (blocks > 512) blocks = 512;
    gemm_bias_kernel<<<blocks, 256, 0, stream>>>(A, lda, W, bias, oF, oB, ldo, M, N, K, mode);
  };

  dim3 recGrid(BB / 16, 2);

  // ---- enc0: xi = ex @ wih^T + bih ; recurrence -> y0, final h -> feats[:,0:128]
  for (int d = 0; d < 2; ++d)
    gemm(ex, 64, wb_e0wih + (size_t)d * G3 * 64, e0bih + d * G3,
         xi + (size_t)d * BT * G3, nullptr, G3, BT, G3, 64, 0);
  gru_rec_kernel<<<recGrid, 128, 0, stream>>>(xi, (long)BT * G3, TT * G3, G3,
      wb_e0whh, e0bhh, y0, 128, 0, feats, 256, 0);

  // ---- enc1: input y0 (K=128); y output unused, final h -> feats[:,128:256]
  for (int d = 0; d < 2; ++d)
    gemm(y0, 128, wb_e1wih + (size_t)d * G3 * 128, e1bih + d * G3,
         xi + (size_t)d * BT * G3, nullptr, G3, BT, G3, 128, 0);
  gru_rec_kernel<<<recGrid, 128, 0, stream>>>(xi, (long)BT * G3, TT * G3, G3,
      wb_e1whh, e1bhh, nullptr, 0, 0, feats, 256, 128);

  // ---- dec0: input is feats broadcast over T -> xi has no t dependence (M=128 GEMM)
  for (int d = 0; d < 2; ++d)
    gemm(feats, 256, wb_d0wih + (size_t)d * G3 * 256, d0bih + d * G3,
         xi_d0 + (size_t)d * BB * G3, nullptr, G3, BB, G3, 256, 0);
  gru_rec_kernel<<<recGrid, 128, 0, stream>>>(xi_d0, (long)BB * G3, G3, /*t-stride*/0,
      wb_d0whh, d0bhh, res, 256, 0, nullptr, 0, 0);

  // ---- dec1: input d0 = res[:,0:128] (lda=256, K=128); output -> res[:,128:256]
  for (int d = 0; d < 2; ++d)
    gemm(res, 256, wb_d1wih + (size_t)d * G3 * 128, d1bih + d * G3,
         xi + (size_t)d * BT * G3, nullptr, G3, BT, G3, 128, 0);
  gru_rec_kernel<<<recGrid, 128, 0, stream>>>(xi, (long)BT * G3, TT * G3, G3,
      wb_d1whh, d1bhh, res, 256, 128, nullptr, 0, 0);

  // ---- fc1: hdd = selu(res @ fc1^T + b) -> bf16
  gemm(res, 256, wb_fc1, fc1b, nullptr, hdd, 64, BT, 64, 256, 1);

  // ---- fc2 + log_softmax + CE fused (no logits materialization)
  fc2_ce_kernel<<<BT / 16, 256, 0, stream>>>(hdd, wb_fc2, fc2b, x, ce);

  // ---- masked mean over t, mean over b -> scalar
  loss_kernel<<<1, 128, 0, stream>>>(ce, x, out);

  (void)in_sizes; (void)n_in; (void)out_size; (void)ws_size;
}
